// ParameterKernel_36112085025350
// MI455X (gfx1250) — matrix operand order repaired and checked by reading, not compiled, not run on hardware
//
#include <hip/hip_runtime.h>
#include <hip/hip_bf16.h>

// ---------------------------------------------------------------------------
// Fused masked polynomial-kernel GEMM chain for MI455X (gfx1250, wave32).
//   S  = X_ref @ desc^T                (4096 x 8192, K=512)
//   K  = mask(Z_ref==Z) * S^2
//   y  = Alpha @ K                     (256 x 8192, K=4096)
// fp32 inputs are split into bf16 hi/lo pairs; each fp32 product is computed
// with 3x v_wmma_f32_16x16x32_bf16 (hi*hi + hi*lo + lo*hi).
// ---------------------------------------------------------------------------

typedef __attribute__((ext_vector_type(16))) __bf16 v16bf;
typedef __attribute__((ext_vector_type(8)))  float  v8f;
typedef __attribute__((ext_vector_type(4)))  unsigned int u32x4;

#define NREF   4096
#define NIN    8192
#define DF     512
#define PDIM   256

#define BJ     64          // output columns per block
#define IC     128         // i rows per chunk
#define KS     256         // k slice staged in LDS
#define ISPLIT 4
#define IPER   (NREF / ISPLIT)   // 1024

#define DSTR   (DF + 8)    // 520  (bf16 elems, padded to dodge bank conflicts)
#define XSTR   (KS + 8)    // 264
#define SSTR   (IC + 8)    // 136

#define SMEM_BYTES ((2 * BJ * DSTR + 2 * IC * XSTR + 2 * BJ * SSTR) * 2)

static __device__ __forceinline__ v8f wmma_bf16(v16bf a, v16bf b, v8f c) {
  return __builtin_amdgcn_wmma_f32_16x16x32_bf16(false, a, false, b,
                                                 (short)0, c, false, false);
}

// A fragment (16x32 bf16, M rows striped in lanes): per lane two contiguous
// 8-element runs: lanes 0-15 -> K[0..7],[16..23]; lanes 16-31 -> K[8..15],[24..31].
static __device__ __forceinline__ v16bf afrag(const __bf16* row, int k0, int half) {
  v16bf v;
  const int g = half << 3;
  ((u32x4*)&v)[0] = *(const u32x4*)(row + k0 + g);
  ((u32x4*)&v)[1] = *(const u32x4*)(row + k0 + 16 + g);
  return v;
}

// B fragment (32x16 bf16, N cols striped in lanes): per lane two contiguous
// 8-element runs: lanes 0-15 -> K[0..7],[16..23]; lanes 16-31 -> K[8..15],[24..31].
static __device__ __forceinline__ v16bf bfrag(const __bf16* row, int k0, int half) {
  v16bf v;
  const __bf16* p = row + k0 + (half << 3);
  ((u32x4*)&v)[0] = *(const u32x4*)p;
  ((u32x4*)&v)[1] = *(const u32x4*)(p + 16);
  return v;
}

static __device__ __forceinline__ unsigned pack_bf16(__bf16 a, __bf16 b) {
  return (unsigned)__builtin_bit_cast(unsigned short, a) |
         ((unsigned)__builtin_bit_cast(unsigned short, b) << 16);
}

// ---------------------------------------------------------------------------
// fp32 -> (bf16 hi, bf16 lo) split
// ---------------------------------------------------------------------------
__global__ void split_bf16_kernel(const float* __restrict__ src,
                                  __bf16* __restrict__ hi,
                                  __bf16* __restrict__ lo, int n) {
  int i = blockIdx.x * blockDim.x + threadIdx.x;
  int stride = gridDim.x * blockDim.x;
  for (; i < n; i += stride) {
    float x = src[i];
    __bf16 h = (__bf16)x;
    hi[i] = h;
    lo[i] = (__bf16)(x - (float)h);
  }
}

// ---------------------------------------------------------------------------
// Fused kernel: grid (NIN/BJ, ISPLIT), block 256 threads (8 wave32)
// ---------------------------------------------------------------------------
__global__ __launch_bounds__(256)
void fused_masked_gemm_kernel(const __bf16* __restrict__ Dhi_g,
                              const __bf16* __restrict__ Dlo_g,
                              const __bf16* __restrict__ Xhi_g,
                              const __bf16* __restrict__ Xlo_g,
                              const __bf16* __restrict__ Ahi_g,
                              const __bf16* __restrict__ Alo_g,
                              const int* __restrict__ Zref,
                              const int* __restrict__ Zin,
                              float* __restrict__ out) {
  extern __shared__ __align__(16) char smem[];
  __bf16* Dh  = (__bf16*)smem;            // [BJ][DSTR]  desc tile (hi)
  __bf16* Dl  = Dh + BJ * DSTR;           // [BJ][DSTR]  desc tile (lo)
  __bf16* Xh  = Dl + BJ * DSTR;           // [IC][XSTR]  X slice  (hi)
  __bf16* Xl  = Xh + IC * XSTR;           // [IC][XSTR]  X slice  (lo)
  __bf16* S2h = Xl + IC * XSTR;           // [BJ][SSTR]  masked S^2 (hi), [j][i]
  __bf16* S2l = S2h + BJ * SSTR;          // [BJ][SSTR]  masked S^2 (lo)

  const int tid  = threadIdx.x;
  const int lane = tid & 31;
  const int w    = tid >> 5;
  const int half = lane >> 4;
  const int ln   = lane & 15;

  const int j0 = blockIdx.x * BJ;
  const int i0 = blockIdx.y * IPER;

  // --- stage desc tile (reused by every i-chunk of this block) -------------
  for (int idx = tid; idx < BJ * (DF / 8); idx += 256) {
    int row = idx / (DF / 8);
    int k   = (idx % (DF / 8)) * 8;
    *(u32x4*)&Dh[row * DSTR + k] = *(const u32x4*)&Dhi_g[(size_t)(j0 + row) * DF + k];
    *(u32x4*)&Dl[row * DSTR + k] = *(const u32x4*)&Dlo_g[(size_t)(j0 + row) * DF + k];
  }

  // element type of this lane's output column, per j-tile
  int zc[BJ / 16];
#pragma unroll
  for (int t = 0; t < BJ / 16; ++t) zc[t] = Zin[j0 + t * 16 + ln];

  // GEMM2 wave tiling of Y^T (BJ x PDIM): row-tile rt (j), 8 col-tiles (p)
  const int rt  = w & 3;
  const int cpb = (w >> 2) * 8;
  v8f Y[8] = {};

  for (int c = 0; c < IPER / IC; ++c) {
    const int ib = i0 + c * IC;
    v8f S[BJ / 16] = {};

    // ---- GEMM1: S strip (16 rows x BJ cols) per wave, K = DF -------------
    for (int kk = 0; kk < DF; kk += KS) {
      __syncthreads();   // prior readers of Xh/Xl (and S2) are done
      for (int idx = tid; idx < IC * (KS / 8); idx += 256) {
        int row = idx / (KS / 8);
        int k   = (idx % (KS / 8)) * 8;
        *(u32x4*)&Xh[row * XSTR + k] =
            *(const u32x4*)&Xhi_g[(size_t)(ib + row) * DF + kk + k];
        *(u32x4*)&Xl[row * XSTR + k] =
            *(const u32x4*)&Xlo_g[(size_t)(ib + row) * DF + kk + k];
      }
      __syncthreads();

      const __bf16* xh_row = Xh + (w * 16 + ln) * XSTR;
      const __bf16* xl_row = Xl + (w * 16 + ln) * XSTR;
      for (int k32 = 0; k32 < KS; k32 += 32) {
        v16bf ah = afrag(xh_row, k32, half);
        v16bf al = afrag(xl_row, k32, half);
#pragma unroll
        for (int t = 0; t < BJ / 16; ++t) {
          const __bf16* dh_row = Dh + (t * 16 + ln) * DSTR;
          const __bf16* dl_row = Dl + (t * 16 + ln) * DSTR;
          v16bf bh = bfrag(dh_row, kk + k32, half);
          v16bf bl = bfrag(dl_row, kk + k32, half);
          S[t] = wmma_bf16(ah, bh, S[t]);
          S[t] = wmma_bf16(ah, bl, S[t]);
          S[t] = wmma_bf16(al, bh, S[t]);
        }
      }
    }

    // ---- square + element mask + re-split, store transposed to LDS -------
    // C layout: VGPR r, lanes 0-15 -> M=r, lanes 16-31 -> M=r+8; N = lane&15
    int zr[8];
#pragma unroll
    for (int r = 0; r < 8; ++r) zr[r] = Zref[ib + w * 16 + 8 * half + r];

#pragma unroll
    for (int t = 0; t < BJ / 16; ++t) {
      const int jl = t * 16 + ln;
#pragma unroll
      for (int r2 = 0; r2 < 4; ++r2) {
        float v0 = S[t][2 * r2], v1 = S[t][2 * r2 + 1];
        v0 *= v0; v1 *= v1;
        if (zr[2 * r2]     != zc[t]) v0 = 0.0f;
        if (zr[2 * r2 + 1] != zc[t]) v1 = 0.0f;
        __bf16 h0 = (__bf16)v0; __bf16 l0 = (__bf16)(v0 - (float)h0);
        __bf16 h1 = (__bf16)v1; __bf16 l1 = (__bf16)(v1 - (float)h1);
        const int il = w * 16 + 8 * half + 2 * r2;   // even -> 4B aligned
        *(unsigned*)&S2h[jl * SSTR + il] = pack_bf16(h0, h1);
        *(unsigned*)&S2l[jl * SSTR + il] = pack_bf16(l0, l1);
      }
    }
    __syncthreads();   // S2 visible to all waves

    // ---- GEMM2: Y^T[j, p] += S2^T[j, i] * Alpha^T[i, p] ------------------
    const __bf16* s2h_row = S2h + (rt * 16 + ln) * SSTR;
    const __bf16* s2l_row = S2l + (rt * 16 + ln) * SSTR;
    for (int k32 = 0; k32 < IC; k32 += 32) {
      v16bf ah = afrag(s2h_row, k32, half);
      v16bf al = afrag(s2l_row, k32, half);
      const size_t kg = (size_t)ib + k32;
#pragma unroll
      for (int u = 0; u < 8; ++u) {
        const int p = (cpb + u) * 16 + ln;
        v16bf bh = bfrag(Ahi_g + (size_t)p * NREF + kg, 0, half);
        v16bf bl = bfrag(Alo_g + (size_t)p * NREF + kg, 0, half);
        Y[u] = wmma_bf16(ah, bh, Y[u]);
        Y[u] = wmma_bf16(ah, bl, Y[u]);
        Y[u] = wmma_bf16(al, bh, Y[u]);
      }
    }
  }

  // ---- combine across ISPLIT partial blocks ------------------------------
#pragma unroll
  for (int u = 0; u < 8; ++u) {
    const int p = (cpb + u) * 16 + ln;
    float* op = out + (size_t)p * NIN + j0 + rt * 16 + 8 * half;
#pragma unroll
    for (int r = 0; r < 8; ++r) unsafeAtomicAdd(op + r, Y[u][r]);
  }
}

// ---------------------------------------------------------------------------
extern "C" void kernel_launch(void* const* d_in, const int* in_sizes, int n_in,
                              void* d_out, int out_size, void* d_ws, size_t ws_size,
                              hipStream_t stream) {
  const float* Alpha = (const float*)d_in[0];   // [256, 4096]
  const float* X_ref = (const float*)d_in[1];   // [4096, 512]
  const float* desc  = (const float*)d_in[2];   // [8192, 512]
  const int*   Z_ref = (const int*)d_in[3];     // [4096]
  const int*   Z     = (const int*)d_in[4];     // [8192]
  // d_in[5] = expK (==2, baked into the square)

  // workspace layout: bf16 hi/lo splits (28 MB total)
  __bf16* ws  = (__bf16*)d_ws;
  __bf16* Dhi = ws;                         // 8192*512
  __bf16* Dlo = Dhi + (size_t)NIN * DF;
  __bf16* Xhi = Dlo + (size_t)NIN * DF;     // 4096*512
  __bf16* Xlo = Xhi + (size_t)NREF * DF;
  __bf16* Ahi = Xlo + (size_t)NREF * DF;    // 256*4096
  __bf16* Alo = Ahi + (size_t)PDIM * NREF;

  hipMemsetAsync(d_out, 0, (size_t)out_size * sizeof(float), stream);

  split_bf16_kernel<<<2048, 256, 0, stream>>>(desc,  Dhi, Dlo, NIN * DF);
  split_bf16_kernel<<<2048, 256, 0, stream>>>(X_ref, Xhi, Xlo, NREF * DF);
  split_bf16_kernel<<<1024, 256, 0, stream>>>(Alpha, Ahi, Alo, PDIM * NREF);

  dim3 grid(NIN / BJ, ISPLIT);
  fused_masked_gemm_kernel<<<grid, 256, SMEM_BYTES, stream>>>(
      Dhi, Dlo, Xhi, Xlo, Ahi, Alo, Z_ref, Z, (float*)d_out);
}
